// MultiHeadAttention_54047868453159
// MI455X (gfx1250) — matrix operand order repaired and checked by reading, not compiled, not run on hardware
//
#include <hip/hip_runtime.h>
#include <hip/hip_bf16.h>

// ---------------------------------------------------------------------------
// MHA forward for MI455X (gfx1250), wave32, bf16 WMMA (16x16x32, f32 accum).
// ---------------------------------------------------------------------------

typedef __attribute__((ext_vector_type(16))) __bf16 v16bf;
typedef __attribute__((ext_vector_type(8)))  float  v8f;
typedef __attribute__((__vector_size__(16))) int    v4i;   // matches async builtin param

#define BB 8
#define SS 2048
#define DM 256
#define HH 4
#define DEPTH 64

#define AS_GLOBAL __attribute__((address_space(1)))
#define AS_LDS    __attribute__((address_space(3)))

#define HAS_ASYNC_LDS (__has_builtin(__builtin_amdgcn_global_load_async_to_lds_b128) && \
                       __has_builtin(__builtin_amdgcn_s_wait_asynccnt))

union AFrag {
    v16bf v;
    unsigned int u[8];
    __bf16 b[16];
};

// Packed f32x2 -> bf16x2 (single v_cvt_pk_bf16_f32 when available).
__device__ __forceinline__ unsigned int pack_bf16(float x, float y) {
#if __has_builtin(__builtin_amdgcn_cvt_pk_bf16_f32)
    auto r = __builtin_amdgcn_cvt_pk_bf16_f32(x, y);
    return __builtin_bit_cast(unsigned int, r);
#else
    union { __bf16 b[2]; unsigned int u; } t;
    t.b[0] = (__bf16)x;
    t.b[1] = (__bf16)y;
    return t.u;
#endif
}

// ---------------------------------------------------------------------------
// Kernel 0a: weights f32 [K][N] -> bf16 transposed [N][K] (B-frags contiguous).
// ---------------------------------------------------------------------------
__global__ void mha_wconv(const float* __restrict__ wq, const float* __restrict__ wk,
                          const float* __restrict__ wv, const float* __restrict__ wo,
                          __bf16* __restrict__ wt) {
    const float* w = (blockIdx.y == 0) ? wq : (blockIdx.y == 1) ? wk
                   : (blockIdx.y == 2) ? wv : wo;
    __bf16* o = wt + blockIdx.y * (DM * DM);
    int n = blockIdx.x, k = threadIdx.x;
    o[n * DM + k] = (__bf16)w[k * DM + n];
}

// ---------------------------------------------------------------------------
// Kernel 0b: activations f32 -> bf16 row-major (conversion hoisted out of GEMM).
// grid = (2048, 3), block 256; each thread converts 8 elems (f32x8 -> bf16x8).
// ---------------------------------------------------------------------------
__global__ __launch_bounds__(256) void mha_xconv(
    const float* __restrict__ q, const float* __restrict__ k, const float* __restrict__ v,
    __bf16* __restrict__ xq, __bf16* __restrict__ xk, __bf16* __restrict__ xv) {
    const float* src = (blockIdx.y == 0) ? q : (blockIdx.y == 1) ? k : v;
    __bf16* dst = (blockIdx.y == 0) ? xq : (blockIdx.y == 1) ? xk : xv;
    size_t i = ((size_t)blockIdx.x * 256 + threadIdx.x) * 8;
    float4 a = *(const float4*)(src + i);
    float4 b = *(const float4*)(src + i + 4);
    uint4 r;
    r.x = pack_bf16(a.x, a.y);
    r.y = pack_bf16(a.z, a.w);
    r.z = pack_bf16(b.x, b.y);
    r.w = pack_bf16(b.z, b.w);
    *(uint4*)(dst + i) = r;
}

// ---------------------------------------------------------------------------
// Kernel 1: fused QKV projection, all-bf16 operands.
// grid = (M/16, 2, 3), block 128.  Each wave: one 16-row strip x 2 col tiles
// (A-fragment gathers amortized over 2 WMMAs).  Writes [B,H,S,DEPTH] bf16.
// ---------------------------------------------------------------------------
__global__ __launch_bounds__(128) void mha_qkv_proj(
    const __bf16* __restrict__ xq, const __bf16* __restrict__ xk, const __bf16* __restrict__ xv,
    const __bf16* __restrict__ wt,
    const float* __restrict__ bq, const float* __restrict__ bk, const float* __restrict__ bv,
    __bf16* __restrict__ qh, __bf16* __restrict__ kh, __bf16* __restrict__ vh) {

    int wave = threadIdx.x >> 5, lane = threadIdx.x & 31;
    int m = lane & 15, half = lane >> 4;
    int mat = blockIdx.z;

    const __bf16* X = (mat == 0) ? xq : (mat == 1) ? xk : xv;
    const __bf16* W = wt + mat * (DM * DM);
    const float* bias = (mat == 0) ? bq : (mat == 1) ? bk : bv;
    __bf16* out = (mat == 0) ? qh : (mat == 1) ? kh : vh;

    int row0 = blockIdx.x * 16;
    int ct0 = (blockIdx.y * 4 + wave) * 2;          // first of 2 col tiles
    int n0 = ct0 * 16 + m;
    int n1 = n0 + 16;

    const __bf16* Arow = X + (size_t)(row0 + m) * DM;
    const __bf16* Wr0 = W + (size_t)n0 * DM;
    const __bf16* Wr1 = W + (size_t)n1 * DM;

    v8f acc0 = {}, acc1 = {};
#pragma unroll
    for (int ks = 0; ks < 8; ++ks) {
        AFrag a;
#pragma unroll
        for (int p = 0; p < 8; ++p) {
            int k0 = ks * 32 + 16 * half + 2 * p;
            a.u[p] = *(const unsigned int*)(Arow + k0);
        }
        v16bf b0 = *(const v16bf*)(Wr0 + ks * 32 + 16 * half);
        v16bf b1 = *(const v16bf*)(Wr1 + ks * 32 + 16 * half);
        acc0 = __builtin_amdgcn_wmma_f32_16x16x32_bf16(false, a.v, false, b0,
                                                       (short)0, acc0, false, false);
        acc1 = __builtin_amdgcn_wmma_f32_16x16x32_bf16(false, a.v, false, b1,
                                                       (short)0, acc1, false, false);
    }

#pragma unroll
    for (int t = 0; t < 2; ++t) {
        int n = (t == 0) ? n0 : n1;
        v8f acc = (t == 0) ? acc0 : acc1;
        float bval = bias[n];
        int hh = n >> 6, d = n & 63;
#pragma unroll
        for (int r = 0; r < 8; ++r) {
            int row = row0 + r + 8 * half;          // D layout: M = r + 8*half
            int b_ = row >> 11, s_ = row & 2047;
            out[(((size_t)(b_ * HH + hh) * SS + s_) << 6) + d] = (__bf16)(acc[r] + bval);
        }
    }
}

// ---------------------------------------------------------------------------
// Kernel 2: flash attention.  grid = (S/64, H, B), block = 128 (4 waves).
// K tile staged to LDS via global_load_async_to_lds_b128 (ASYNCcnt);
// V transposed into LDS; P goes D-layout -> A-layout through LDS.
// ---------------------------------------------------------------------------
__global__ __launch_bounds__(128) void mha_attn(
    const __bf16* __restrict__ qh, const __bf16* __restrict__ kh,
    const __bf16* __restrict__ vh, const float* __restrict__ mask,
    __bf16* __restrict__ concat) {

    __shared__ __align__(32) __bf16 Kl[64][DEPTH];     // K tile, row-major [kc][d]
    __shared__ __align__(32) __bf16 Vt[DEPTH][64];     // V^T tile [d][kc]
    __shared__ __align__(32) __bf16 Pl[4][16][64];     // per-wave P tile

    int wave = threadIdx.x >> 5, lane = threadIdx.x & 31;
    int m = lane & 15, half = lane >> 4;
    int qt = blockIdx.x, h = blockIdx.y, b = blockIdx.z;

    const __bf16* Qp = qh + (size_t)(b * HH + h) * SS * DEPTH;
    const __bf16* Kp = kh + (size_t)(b * HH + h) * SS * DEPTH;
    const __bf16* Vp = vh + (size_t)(b * HH + h) * SS * DEPTH;
    const float* mp = mask + (size_t)b * SS;

    // Q fragments held in registers; pre-scaled by 1/sqrt(depth) = 0.125.
    AFrag qf[2];
    int qrow = qt * 64 + wave * 16 + m;
#pragma unroll
    for (int ks = 0; ks < 2; ++ks) {
#pragma unroll
        for (int p = 0; p < 8; ++p) {
            int k0 = ks * 32 + 16 * half + 2 * p;
            union { unsigned int u; __bf16 b[2]; } t;
            t.u = *(const unsigned int*)(Qp + (size_t)qrow * DEPTH + k0);
            qf[ks].u[p] = pack_bf16((float)t.b[0] * 0.125f, (float)t.b[1] * 0.125f);
        }
    }

    v8f zero = {};
    v8f o[4] = {zero, zero, zero, zero};
    float mi[8], li[8];
#pragma unroll
    for (int r = 0; r < 8; ++r) { mi[r] = -1e30f; li[r] = 0.f; }

    for (int kt = 0; kt < SS / 64; ++kt) {
        __syncthreads();                       // protect LDS readers of prev iter

        // ---- stage K tile (row-major, no transpose needed) ----
        {
            const __bf16* gsrc = Kp + (size_t)kt * 64 * DEPTH + threadIdx.x * 32;
            __bf16* ldst = &Kl[0][0] + threadIdx.x * 32;
#if HAS_ASYNC_LDS
#pragma unroll
            for (int j = 0; j < 4; ++j) {
                __builtin_amdgcn_global_load_async_to_lds_b128(
                    (AS_GLOBAL v4i*)(gsrc + j * 8),
                    (AS_LDS v4i*)(ldst + j * 8), 0, 0);
            }
#else
#pragma unroll
            for (int j = 0; j < 4; ++j)
                *(uint4*)(ldst + j * 8) = *(const uint4*)(gsrc + j * 8);
#endif
        }

        // ---- stage V^T into LDS: thread -> (kc, half of d) ----
        {
            int kc = threadIdx.x & 63, dh = threadIdx.x >> 6;
            const __bf16* src = Vp + (size_t)(kt * 64 + kc) * DEPTH + dh * 32;
#pragma unroll
            for (int j = 0; j < 32; ++j) Vt[dh * 32 + j][kc] = src[j];
        }

#if HAS_ASYNC_LDS
        __builtin_amdgcn_s_wait_asynccnt(0);
#endif
        __syncthreads();

        if (kt + 1 < SS / 64) {
            __builtin_prefetch(Kp + (size_t)(kt + 1) * 64 * DEPTH + threadIdx.x * 32, 0, 1);
            __builtin_prefetch(Vp + (size_t)(kt + 1) * 64 * DEPTH + threadIdx.x * 32, 0, 1);
        }

        // ---- logits S = (Q*scale) K^T : 4 col tiles x 2 k-steps ----
        v8f s[4] = {zero, zero, zero, zero};
#pragma unroll
        for (int c = 0; c < 4; ++c) {
#pragma unroll
            for (int ks = 0; ks < 2; ++ks) {
                v16bf kb = *(const v16bf*)&Kl[c * 16 + m][ks * 32 + 16 * half];
                s[c] = __builtin_amdgcn_wmma_f32_16x16x32_bf16(false, qf[ks].v, false, kb,
                                                               (short)0, s[c], false, false);
            }
        }

        // ---- mask + online softmax (f32) ----
        float sf[4][8];
#pragma unroll
        for (int c = 0; c < 4; ++c) {
            float mk = -1e9f * mp[kt * 64 + c * 16 + m];
#pragma unroll
            for (int r = 0; r < 8; ++r) sf[c][r] = s[c][r] + mk;
        }
        float mnew[8], alpha[8];
#pragma unroll
        for (int r = 0; r < 8; ++r) {
            float mx = fmaxf(fmaxf(sf[0][r], sf[1][r]), fmaxf(sf[2][r], sf[3][r]));
            mx = fmaxf(mx, __shfl_xor(mx, 1, 32));
            mx = fmaxf(mx, __shfl_xor(mx, 2, 32));
            mx = fmaxf(mx, __shfl_xor(mx, 4, 32));
            mx = fmaxf(mx, __shfl_xor(mx, 8, 32));
            mnew[r] = fmaxf(mi[r], mx);
            alpha[r] = __expf(mi[r] - mnew[r]);
            mi[r] = mnew[r];
        }
        float rs[8];
#pragma unroll
        for (int r = 0; r < 8; ++r) rs[r] = 0.f;
#pragma unroll
        for (int c = 0; c < 4; ++c)
#pragma unroll
            for (int r = 0; r < 8; ++r) {
                float p = __expf(sf[c][r] - mnew[r]);
                sf[c][r] = p;
                rs[r] += p;
            }
#pragma unroll
        for (int r = 0; r < 8; ++r) {
            float t = rs[r];
            t += __shfl_xor(t, 1, 32);
            t += __shfl_xor(t, 2, 32);
            t += __shfl_xor(t, 4, 32);
            t += __shfl_xor(t, 8, 32);
            li[r] = li[r] * alpha[r] + t;
#pragma unroll
            for (int c = 0; c < 4; ++c) o[c][r] *= alpha[r];
        }

        // ---- P: D-layout -> LDS -> A-layout ----
#pragma unroll
        for (int c = 0; c < 4; ++c)
#pragma unroll
            for (int r = 0; r < 8; ++r)
                Pl[wave][r + 8 * half][c * 16 + m] = (__bf16)sf[c][r];

        // ---- O += P V ----
#pragma unroll
        for (int ks = 0; ks < 2; ++ks) {
            AFrag pf;
#pragma unroll
            for (int p = 0; p < 8; ++p) {
                int k0 = ks * 32 + 16 * half + 2 * p;
                pf.u[p] = *(const unsigned int*)&Pl[wave][m][k0];
            }
#pragma unroll
            for (int c = 0; c < 4; ++c) {
                v16bf vb = *(const v16bf*)&Vt[c * 16 + m][ks * 32 + 16 * half];
                o[c] = __builtin_amdgcn_wmma_f32_16x16x32_bf16(false, pf.v, false, vb,
                                                               (short)0, o[c], false, false);
            }
        }
    }

    // ---- normalize and write concat [B,S,DM] bf16 ----
#pragma unroll
    for (int r = 0; r < 8; ++r) {
        float inv = 1.0f / li[r];
        int srow = qt * 64 + wave * 16 + r + 8 * half;
#pragma unroll
        for (int c = 0; c < 4; ++c) {
            int col = h * 64 + c * 16 + m;
            concat[((size_t)b * SS + srow) * DM + col] = (__bf16)(o[c][r] * inv);
        }
    }
}

// ---------------------------------------------------------------------------
// Kernel 3: output projection concat(bf16) @ Wo^T + bo -> f32 out.
// grid = (M/16, 2), block 128; 2 col tiles per wave.
// ---------------------------------------------------------------------------
__global__ __launch_bounds__(128) void mha_out_proj(
    const __bf16* __restrict__ concat, const __bf16* __restrict__ wot,
    const float* __restrict__ bo, float* __restrict__ out) {

    int wave = threadIdx.x >> 5, lane = threadIdx.x & 31;
    int m = lane & 15, half = lane >> 4;
    int row0 = blockIdx.x * 16;
    int ct0 = (blockIdx.y * 4 + wave) * 2;
    int n0 = ct0 * 16 + m;
    int n1 = n0 + 16;

    const __bf16* Arow = concat + (size_t)(row0 + m) * DM;
    const __bf16* Wr0 = wot + (size_t)n0 * DM;
    const __bf16* Wr1 = wot + (size_t)n1 * DM;

    v8f acc0 = {}, acc1 = {};
#pragma unroll
    for (int ks = 0; ks < 8; ++ks) {
        AFrag a;
#pragma unroll
        for (int p = 0; p < 8; ++p) {
            int k0 = ks * 32 + 16 * half + 2 * p;
            a.u[p] = *(const unsigned int*)(Arow + k0);
        }
        v16bf b0 = *(const v16bf*)(Wr0 + ks * 32 + 16 * half);
        v16bf b1 = *(const v16bf*)(Wr1 + ks * 32 + 16 * half);
        acc0 = __builtin_amdgcn_wmma_f32_16x16x32_bf16(false, a.v, false, b0,
                                                       (short)0, acc0, false, false);
        acc1 = __builtin_amdgcn_wmma_f32_16x16x32_bf16(false, a.v, false, b1,
                                                       (short)0, acc1, false, false);
    }

    float bv0 = bo[n0], bv1 = bo[n1];
#pragma unroll
    for (int r = 0; r < 8; ++r) {
        int row = row0 + r + 8 * half;
        out[(size_t)row * DM + n0] = acc0[r] + bv0;
        out[(size_t)row * DM + n1] = acc1[r] + bv1;
    }
}

// ---------------------------------------------------------------------------
extern "C" void kernel_launch(void* const* d_in, const int* in_sizes, int n_in,
                              void* d_out, int out_size, void* d_ws, size_t ws_size,
                              hipStream_t stream) {
    const float* v    = (const float*)d_in[0];
    const float* k    = (const float*)d_in[1];
    const float* q    = (const float*)d_in[2];
    const float* mask = (const float*)d_in[3];
    const float* wq   = (const float*)d_in[4];
    const float* bq   = (const float*)d_in[5];
    const float* wk   = (const float*)d_in[6];
    const float* bk   = (const float*)d_in[7];
    const float* wv   = (const float*)d_in[8];
    const float* bv   = (const float*)d_in[9];
    const float* wo   = (const float*)d_in[10];
    const float* bo   = (const float*)d_in[11];
    float* out = (float*)d_out;

    const size_t NTOK = (size_t)BB * SS;          // 16384 rows
    __bf16* wt     = (__bf16*)d_ws;               // 4 * 256*256
    __bf16* xq     = wt + 4 * DM * DM;            // NTOK*DM each
    __bf16* xk     = xq + NTOK * DM;
    __bf16* xv     = xk + NTOK * DM;
    __bf16* qhp    = xv + NTOK * DM;              // B*H*S*DEPTH each (== NTOK*DM)
    __bf16* khp    = qhp + NTOK * DM;
    __bf16* vhp    = khp + NTOK * DM;
    __bf16* concat = vhp + NTOK * DM;

    mha_wconv   <<<dim3(DM, 4),                  DM,  0, stream>>>(wq, wk, wv, wo, wt);
    mha_xconv   <<<dim3(NTOK * DM / 2048, 3),    256, 0, stream>>>(q, k, v, xq, xk, xv);
    mha_qkv_proj<<<dim3(NTOK / 16, 2, 3),        128, 0, stream>>>(xq, xk, xv, wt, bq, bk, bv,
                                                                   qhp, khp, vhp);
    mha_attn    <<<dim3(SS / 64, HH, BB),        128, 0, stream>>>(qhp, khp, vhp, mask, concat);
    mha_out_proj<<<dim3(NTOK / 16, 2),           128, 0, stream>>>(concat, wt + 3 * DM * DM,
                                                                   bo, out);
}
